// GroupedQueryAttention_71502615544037
// MI455X (gfx1250) — compile-verified, matrix operand order checked by reading, not run on hardware
//
#include <hip/hip_runtime.h>

typedef unsigned short u16;
typedef unsigned int   u32;
typedef unsigned long long u64;
typedef __attribute__((ext_vector_type(4)))  int    v4i;
typedef __attribute__((ext_vector_type(16))) __bf16 v16bf;
typedef __attribute__((ext_vector_type(8)))  float  v8f;

// ---------- CDNA5 async global->LDS staging ----------

#if defined(__has_builtin)
#if __has_builtin(__builtin_amdgcn_global_load_async_to_lds_b128)
#define ASYNC_LDS 1
#endif
#endif

__device__ __forceinline__ void async_b128(const u16* g, u16* l) {
#ifdef ASYNC_LDS
  // Toolchain signature (from diagnostics): generic v4i* params, imm offset, imm cpol.
  __builtin_amdgcn_global_load_async_to_lds_b128((v4i*)(u64)g, (v4i*)l, 0, 0);
#else
  *(uint4*)l = *(const uint4*)g;   // sync fallback through VGPRs
#endif
}

__device__ __forceinline__ void wait_async0() {
#ifdef ASYNC_LDS
#if __has_builtin(__builtin_amdgcn_s_wait_asynccnt)
  __builtin_amdgcn_s_wait_asynccnt(0);
#else
  asm volatile("s_wait_asynccnt 0x0" ::: "memory");
#endif
#else
  asm volatile("s_wait_dscnt 0x0" ::: "memory");
#endif
}

// ---------- helpers ----------

__device__ __forceinline__ u16 f2bf(float f) {
  u32 u = __builtin_bit_cast(u32, f);
  u32 r = u + 0x7FFFu + ((u >> 16) & 1u);   // round-to-nearest-even
  return (u16)(r >> 16);
}

union FragU { uint4 u[2]; v16bf v; };

// A/B fragment for v_wmma_*_16x16x32: lane holds 8 halfs at koff and 8 at koff+16,
// koff = (lane>=16 ? 8 : 0) within the 32-wide K chunk selected by kc.
__device__ __forceinline__ v16bf load_frag(const u16* rowbase, int lane, int kc) {
  const int koff = kc * 32 + ((lane >> 4) << 3);
  FragU f;
  f.u[0] = *(const uint4*)(rowbase + koff);
  f.u[1] = *(const uint4*)(rowbase + koff + 16);
  return f.v;
}

__device__ __forceinline__ v8f wmma_bf16(v16bf a, v16bf b, v8f c) {
  return __builtin_amdgcn_wmma_f32_16x16x32_bf16(false, a, false, b, (short)0, c, false, false);
}

// ---------- prep kernels ----------

__global__ void prep_x(const float* __restrict__ x, u16* __restrict__ xb, int n4) {
  int i = blockIdx.x * blockDim.x + threadIdx.x;
  if (i >= n4) return;
  float4 v = ((const float4*)x)[i];
  ushort4 p;
  p.x = f2bf(v.x); p.y = f2bf(v.y); p.z = f2bf(v.z); p.w = f2bf(v.w);
  ((ushort4*)xb)[i] = p;
}

// Transposed bf16 weights [N=3072][K=4096]:
// rows 0..1023: wq_eff^T (group-summed, pre-scaled 1/sqrt(128)); 1024..2047: wk^T; 2048..3071: wv^T
__global__ void prep_wqkv(const float* __restrict__ wq, const float* __restrict__ wk,
                          const float* __restrict__ wv, u16* __restrict__ wT) {
  size_t idx = (size_t)blockIdx.x * blockDim.x + threadIdx.x;
  int n = (int)(idx >> 12);
  int k = (int)(idx & 4095);
  float val;
  if (n < 1024) {
    int hh = n >> 7, d = n & 127;
    const float* base = wq + (size_t)k * 4096 + hh * 512 + d;
    val = (base[0] + base[128] + base[256] + base[384]) * 0.08838834764831845f;
  } else if (n < 2048) {
    val = wk[(size_t)k * 1024 + (n - 1024)];
  } else {
    val = wv[(size_t)k * 1024 + (n - 2048)];
  }
  wT[idx] = f2bf(val);
}

__global__ void prep_wo(const float* __restrict__ wo, u16* __restrict__ woT) {
  size_t idx = (size_t)blockIdx.x * blockDim.x + threadIdx.x;
  int n = (int)(idx >> 10);
  int k = (int)(idx & 1023);
  woT[idx] = f2bf(wo[(size_t)k * 4096 + n]);
}

// ---------- double-buffered bf16 GEMM: C[M][N] = A[M][K] * Bt[N][K]^T ----------
// Block: 256 threads / 8 waves. Block tile 256(M) x 128(N), K-step 64.
// Wave w owns rows w*32..w*32+32, all 128 N cols (2 m-tiles x 8 n-tiles of WMMA).
// Dynamic LDS, async global->LDS staging, double buffered.
// For the QKV GEMM, columns >= 2048 (the V projection) are written TRANSPOSED
// into VT[b][h*128+d][s] so flash attention can stage V without transposing.

#define LDT 72   // LDS row stride in halfs (64 + 8 pad)
#define ABUF (256 * LDT)
#define BBUF (128 * LDT)
#define BUFSTRIDE (ABUF + BBUF)

template<bool OUT_BF16>
__global__ __launch_bounds__(256)
void gemm_bf16_nt(const u16* __restrict__ A, const u16* __restrict__ Bt,
                  void* __restrict__ C, u16* __restrict__ VT,
                  int M, int N, int K) {
  extern __shared__ u16 smem[];

  const int tid  = threadIdx.x;
  const int lane = tid & 31;
  const int w    = tid >> 5;
  const int m0   = blockIdx.y * 256;
  const int n0   = blockIdx.x * 128;
  const int lrow = lane & 15;

  v8f zero = {0.f, 0.f, 0.f, 0.f, 0.f, 0.f, 0.f, 0.f};
  v8f acc[2][8];
#pragma unroll
  for (int mt = 0; mt < 2; ++mt)
#pragma unroll
    for (int nt = 0; nt < 8; ++nt) acc[mt][nt] = zero;

  // cooperative staging mapping
  const u16* agBase = A  + (size_t)(m0 + tid) * K;              // 1 row, 64 cols
  const u16* bgBase = Bt + (size_t)(n0 + (tid >> 1)) * K + (tid & 1) * 32;  // half row
  const int  alOff  = tid * LDT;
  const int  blOff  = (tid >> 1) * LDT + (tid & 1) * 32;

  // prologue: stage buffer 0
  {
    u16* al = smem + alOff;
    u16* bl = smem + ABUF + blOff;
#pragma unroll
    for (int i = 0; i < 8; ++i) async_b128(agBase + i * 8, al + i * 8);
#pragma unroll
    for (int i = 0; i < 4; ++i) async_b128(bgBase + i * 8, bl + i * 8);
  }
  wait_async0();
  __syncthreads();

  for (int k0 = 0; k0 < K; k0 += 64) {
    const int cur = (k0 >> 6) & 1;
    const int nxt = cur ^ 1;
    u16* Acur = smem + cur * BUFSTRIDE;
    u16* Bcur = Acur + ABUF;
    if (k0 + 64 < K) {
      const u16* ag = agBase + k0 + 64;
      const u16* bg = bgBase + k0 + 64;
      u16* al = smem + nxt * BUFSTRIDE + alOff;
      u16* bl = smem + nxt * BUFSTRIDE + ABUF + blOff;
#pragma unroll
      for (int i = 0; i < 8; ++i) async_b128(ag + i * 8, al + i * 8);
#pragma unroll
      for (int i = 0; i < 4; ++i) async_b128(bg + i * 8, bl + i * 8);
    }
#pragma unroll
    for (int kc = 0; kc < 2; ++kc) {
      v16bf af0 = load_frag(&Acur[(w * 32 +      lrow) * LDT], lane, kc);
      v16bf af1 = load_frag(&Acur[(w * 32 + 16 + lrow) * LDT], lane, kc);
#pragma unroll
      for (int nt = 0; nt < 8; ++nt) {
        v16bf bfm = load_frag(&Bcur[(nt * 16 + lrow) * LDT], lane, kc);
        acc[0][nt] = wmma_bf16(af0, bfm, acc[0][nt]);
        acc[1][nt] = wmma_bf16(af1, bfm, acc[1][nt]);
      }
    }
    wait_async0();
    __syncthreads();
  }

  const int msel = (lane >> 4) * 8;
#pragma unroll
  for (int mt = 0; mt < 2; ++mt)
#pragma unroll
    for (int nt = 0; nt < 8; ++nt) {
      int gm = m0 + w * 32 + mt * 16 + msel;
      int gn = n0 + nt * 16 + lrow;
#pragma unroll
      for (int r = 0; r < 8; ++r) {
        float val = acc[mt][nt][r];
        if (OUT_BF16) {
          if (VT != nullptr && gn >= 2048) {
            int hd = gn - 2048;            // h*128 + d
            int brow = gm + r;             // b*2048 + s
            VT[(((size_t)(brow >> 11) << 10) + hd) * 2048 + (brow & 2047)] = f2bf(val);
          } else {
            ((u16*)C)[(size_t)(gm + r) * N + gn] = f2bf(val);
          }
        } else {
          ((float*)C)[(size_t)(gm + r) * N + gn] = val;
        }
      }
    }
}

// ---------- flash attention ----------
// qkv: [4096 = b*2048+s][3072] bf16 (q_eff cols 0..1023 pre-scaled, k cols 1024..2047).
// vT : [b][h*128+d][2048] bf16.  o: [4096][1024] bf16.
// Grid (16 qblocks, 8 heads, 2 batch); 256 thr / 8 waves; wave owns 16 q rows.

#define KLD 136  // K-tile row stride (128 d + 8)
#define VLD 72   // V^T / P row stride (64 keys + 8)

__global__ __launch_bounds__(256)
void flash_gqa(const u16* __restrict__ qkv, const u16* __restrict__ vT,
               u16* __restrict__ o) {
  __shared__ u16 Kt[64 * KLD];        // [key][d]
  __shared__ u16 Vt[128 * VLD];       // [d][key]
  __shared__ u16 Pl[8 * 16 * VLD];    // per-wave P scratch [row][key]

  const int tid  = threadIdx.x;
  const int lane = tid & 31;
  const int w    = tid >> 5;
  const int lrow = lane & 15;
  const int hsel = lane >> 4;
  const int h    = blockIdx.y;
  const size_t rowbase = (size_t)blockIdx.z * 2048;
  const int q0 = blockIdx.x * 128;

  // Preload this wave's Q tile (16 rows x 128 d) as 4 A-fragments.
  v16bf qf[4];
  {
    const u16* qrow = qkv + (rowbase + q0 + w * 16 + lrow) * 3072 + h * 128;
#pragma unroll
    for (int kc = 0; kc < 4; ++kc) qf[kc] = load_frag(qrow, lane, kc);
  }

  v8f zero = {0.f, 0.f, 0.f, 0.f, 0.f, 0.f, 0.f, 0.f};
  v8f Oacc[8];
#pragma unroll
  for (int dt = 0; dt < 8; ++dt) Oacc[dt] = zero;
  float m[8], l[8];
#pragma unroll
  for (int r = 0; r < 8; ++r) { m[r] = -1e30f; l[r] = 0.f; }

  u16* Pw = Pl + w * 16 * VLD;
  // cooperative staging mappings
  const int key = tid >> 2, q4 = tid & 3;      // K tile: 64 keys x 128 d
  const int drow = tid >> 1, half = tid & 1;   // V tile: 128 d x 64 keys
  const u16* vbase = vT + ((size_t)blockIdx.z * 1024 + h * 128 + drow) * 2048 + half * 32;

  for (int j = 0; j < 2048; j += 64) {
    {
      const u16* gk = qkv + (rowbase + j + key) * 3072 + 1024 + h * 128 + q4 * 32;
      u16* lk = Kt + key * KLD + q4 * 32;
#pragma unroll
      for (int i = 0; i < 4; ++i) async_b128(gk + i * 8, lk + i * 8);
      const u16* gv = vbase + j;
      u16* lv = Vt + drow * VLD + half * 32;
#pragma unroll
      for (int i = 0; i < 4; ++i) async_b128(gv + i * 8, lv + i * 8);
    }
    wait_async0();
    __syncthreads();

    // S = Q * K^T : 4 n-tiles of 16 keys, K-dim = d = 128 (4 chunks).
    v8f s[4];
#pragma unroll
    for (int nt = 0; nt < 4; ++nt) {
      s[nt] = zero;
#pragma unroll
      for (int kc = 0; kc < 4; ++kc) {
        v16bf bK = load_frag(&Kt[(nt * 16 + lrow) * KLD], lane, kc);
        s[nt] = wmma_bf16(qf[kc], bK, s[nt]);
      }
    }

    // Online softmax; element r belongs to row (r + hsel*8); reduce across 16 lanes.
    float mx[8], rs[8], rf[8];
#pragma unroll
    for (int r = 0; r < 8; ++r)
      mx[r] = fmaxf(fmaxf(s[0][r], s[1][r]), fmaxf(s[2][r], s[3][r]));
#pragma unroll
    for (int d = 1; d < 16; d <<= 1)
#pragma unroll
      for (int r = 0; r < 8; ++r) mx[r] = fmaxf(mx[r], __shfl_xor(mx[r], d, 16));
#pragma unroll
    for (int r = 0; r < 8; ++r) {
      float mn = fmaxf(m[r], mx[r]);
      rf[r] = __expf(m[r] - mn);
      m[r] = mn;
      rs[r] = 0.f;
    }
#pragma unroll
    for (int nt = 0; nt < 4; ++nt)
#pragma unroll
      for (int r = 0; r < 8; ++r) {
        float p = __expf(s[nt][r] - m[r]);
        rs[r] += p;
        Pw[(r + hsel * 8) * VLD + nt * 16 + lrow] = f2bf(p);
      }
#pragma unroll
    for (int d = 1; d < 16; d <<= 1)
#pragma unroll
      for (int r = 0; r < 8; ++r) rs[r] += __shfl_xor(rs[r], d, 16);
#pragma unroll
    for (int r = 0; r < 8; ++r) l[r] = l[r] * rf[r] + rs[r];

    // Intra-wave LDS RAW: P scatter writes -> P fragment reads.
    asm volatile("s_wait_dscnt 0x0" ::: "memory");

    // O = O*rf + P * V : K-dim = 64 keys (2 chunks), 8 d-tiles.
    v16bf aP[2];
#pragma unroll
    for (int kc = 0; kc < 2; ++kc) aP[kc] = load_frag(&Pw[lrow * VLD], lane, kc);
#pragma unroll
    for (int dt = 0; dt < 8; ++dt) {
#pragma unroll
      for (int r = 0; r < 8; ++r) Oacc[dt][r] *= rf[r];
#pragma unroll
      for (int kc = 0; kc < 2; ++kc) {
        v16bf bV = load_frag(&Vt[(dt * 16 + lrow) * VLD], lane, kc);
        Oacc[dt] = wmma_bf16(aP[kc], bV, Oacc[dt]);
      }
    }
    __syncthreads();
  }

  float rl[8];
#pragma unroll
  for (int r = 0; r < 8; ++r) rl[r] = 1.f / l[r];
#pragma unroll
  for (int dt = 0; dt < 8; ++dt)
#pragma unroll
    for (int r = 0; r < 8; ++r)
      o[(rowbase + q0 + w * 16 + r + hsel * 8) * 1024 + h * 128 + dt * 16 + lrow] =
          f2bf(Oacc[dt][r] * rl[r]);
}

// ---------- launcher ----------

extern "C" void kernel_launch(void* const* d_in, const int* in_sizes, int n_in,
                              void* d_out, int out_size, void* d_ws, size_t ws_size,
                              hipStream_t stream) {
  const float* x  = (const float*)d_in[0];
  const float* wq = (const float*)d_in[1];
  const float* wk = (const float*)d_in[2];
  const float* wv = (const float*)d_in[3];
  const float* wo = (const float*)d_in[4];

  u16* xb  = (u16*)d_ws;                     // [4096][4096]
  u16* wT  = xb  + (size_t)4096 * 4096;      // [3072][4096]
  u16* woT = wT  + (size_t)3072 * 4096;      // [4096][1024]
  u16* qkv = woT + (size_t)4096 * 1024;      // [4096][3072] (v region unused)
  u16* ov  = qkv + (size_t)4096 * 3072;      // [4096][1024]
  u16* vTb = ov  + (size_t)4096 * 1024;      // [2][1024][2048]

  prep_x   <<<(4096 * 4096 / 4) / 256, 256, 0, stream>>>(x, xb, 4096 * 4096 / 4);
  prep_wqkv<<<(3072 * 4096) / 256,     256, 0, stream>>>(wq, wk, wv, wT);
  prep_wo  <<<(4096 * 1024) / 256,     256, 0, stream>>>(wo, woT);

  const u32 gemm_lds = 2 * BUFSTRIDE * (u32)sizeof(u16);  // 110,592 B

  // Fused QKV projection (V written transposed into vTb).
  gemm_bf16_nt<true><<<dim3(3072 / 128, 4096 / 256), 256, gemm_lds, stream>>>(
      xb, wT, qkv, vTb, 4096, 3072, 4096);

  // Flash GQA attention.
  flash_gqa<<<dim3(16, 8, 2), 256, 0, stream>>>(qkv, vTb, ov);

  // Output projection -> fp32 d_out.
  gemm_bf16_nt<false><<<dim3(4096 / 128, 4096 / 256), 256, gemm_lds, stream>>>(
      ov, woT, d_out, nullptr, 4096, 4096, 1024);
}
